// LoCIFusionModule3D_23295902614252
// MI455X (gfx1250) — matrix operand order repaired and checked by reading, not compiled, not run on hardware
//
#include <hip/hip_runtime.h>

// ---------------------------------------------------------------------------
// Types for CDNA5 WMMA (wave32, gfx1250)
// ---------------------------------------------------------------------------
typedef __attribute__((ext_vector_type(16))) __bf16    v16bf;
typedef __attribute__((ext_vector_type(8)))  float     v8f;
typedef __attribute__((ext_vector_type(4)))  unsigned  su4;
typedef __attribute__((ext_vector_type(8)))  unsigned  su8;

__device__ __forceinline__ v8f wmma_bf16(v16bf a, v16bf b, v8f c) {
  // (neg_a, A, neg_b, B, c_mod, C, reuse_a, reuse_b)
  return __builtin_amdgcn_wmma_f32_16x16x32_bf16(false, a, false, b, (short)0, c,
                                                 false, false);
}

__device__ __forceinline__ __bf16 to_bf16(float f) {
  unsigned u = __builtin_bit_cast(unsigned, f);
  unsigned r = u + 0x7FFFu + ((u >> 16) & 1u);   // round-to-nearest-even
  unsigned short h = (unsigned short)(r >> 16);
  return __builtin_bit_cast(__bf16, h);
}
__device__ __forceinline__ void store_val(float* p, float v)  { *p = v; }
__device__ __forceinline__ void store_val(__bf16* p, float v) { *p = to_bf16(v); }

// ---------------------------------------------------------------------------
// CDNA5 async memory->LDS copy (GLOBAL_LOAD_ASYNC_TO_LDS_B128, ASYNCcnt)
// LDS destination address = low 32 bits of the flat shared-aperture address.
// ---------------------------------------------------------------------------
__device__ __forceinline__ void async_copy16(void* lds_dst, const void* gsrc) {
  unsigned l = (unsigned)(unsigned long long)lds_dst;
  unsigned long long g = (unsigned long long)gsrc;
  asm volatile("global_load_async_to_lds_b128 %0, %1, off"
               :
               : "v"(l), "v"(g)
               : "memory");
}
__device__ __forceinline__ void wait_async0() {
  asm volatile("s_wait_asynccnt 0x0" ::: "memory");
}
__device__ __forceinline__ void wait_tensor0() {
#if __has_builtin(__builtin_amdgcn_s_wait_tensorcnt)
  __builtin_amdgcn_s_wait_tensorcnt((short)0);
#else
  asm volatile("s_wait_tensorcnt 0x0" ::: "memory");
#endif
}

// ---------------------------------------------------------------------------
// CDNA5 Tensor Data Mover: 2D tile DMA global->LDS (TENSORcnt).
// Builds D# groups 0/1 per ISA 8.3/8.4 (count=1, type=2 "image",
// 2-byte elements, optional LDS row padding) and issues TENSOR_LOAD_TO_LDS.
// Issue from a single wave; other waves synchronize via the WG barrier.
// ---------------------------------------------------------------------------
__device__ __forceinline__ void tdm_load_2d(void* lds_dst, const void* gsrc,
                                            unsigned tile_w,   // elems per row
                                            unsigned tile_h,   // rows
                                            unsigned stride,   // row stride, elems
                                            unsigned pad_en,   // 1 = pad LDS rows
                                            unsigned pad_ivl,  // 3 -> every 16 DW
                                            unsigned pad_amt)  // 3 -> 4 DW pad
{
  const unsigned long long ga = (unsigned long long)gsrc;
  su4 g0;
  g0[0] = 1u;                                               // count=1, user D#
  g0[1] = (unsigned)(unsigned long long)lds_dst;            // lds_addr
  g0[2] = (unsigned)ga;                                     // global_addr lo
  g0[3] = (unsigned)((ga >> 32) & 0x01FFFFFFu) | (2u << 30);  // addr hi | type=2

  const unsigned td0 = stride;        // tensor_dim0 (no OOB in our tiles)
  const unsigned td1 = 0x7FFFFFFFu;   // tensor_dim1 (large)
  su8 g1;
  g1[0] = (1u << 16) | (pad_en << 20) | (pad_ivl << 22) | (pad_amt << 25);
  g1[1] = (td0 & 0xFFFFu) << 16;                            // dim0[15:0]
  g1[2] = ((td0 >> 16) & 0xFFFFu) | ((td1 & 0xFFFFu) << 16);  // dim0 hi | dim1 lo
  g1[3] = ((td1 >> 16) & 0xFFFFu) | (tile_w << 16);         // dim1 hi | tile_dim0
  g1[4] = tile_h & 0xFFFFu;                                 // tile_dim1 (dim2=0)
  g1[5] = stride;                                           // dim0_stride[31:0]
  g1[6] = 0u;                                               // stride hi | dim1_stride lo
  g1[7] = 0u;
  asm volatile("tensor_load_to_lds %0, %1" :: "s"(g0), "s"(g1) : "memory");
}

// ---------------------------------------------------------------------------
// Fragment loaders per CDNA5 ISA 7.12.2 (16-bit matrices, wave32)
// ---------------------------------------------------------------------------
__device__ __forceinline__ int a_kofs(int v, int half) {
  return (v < 4) ? (v * 2 + half * 8) : (16 + (v - 4) * 2 + half * 8);
}
__device__ __forceinline__ int b_kofs(int v, int half) {
  return (v < 4) ? (v * 2 + half * 8) : (16 + (v - 4) * 2 + half * 8);
}

__device__ __forceinline__ v16bf load_a_frag(const __bf16* tile, int lda,
                                             int m0, int k0, int lane) {
  const int m = m0 + (lane & 15);
  const int h = lane >> 4;
  v16bf f;
#pragma unroll
  for (int v = 0; v < 8; ++v) {
    const int k = k0 + a_kofs(v, h);
    f[2 * v]     = tile[m * lda + k];
    f[2 * v + 1] = tile[m * lda + k + 1];
  }
  return f;
}

// B[k][n] where LDS holds tile[n][k] (K contiguous -> ds_load_b128)
__device__ __forceinline__ v16bf load_b_frag_nk(const __bf16* tile, int ldt,
                                                int k0, int n0, int lane) {
  const int n = n0 + (lane & 15);
  const int h = lane >> 4;
  v16bf f;
#pragma unroll
  for (int v = 0; v < 8; ++v) {
    const int k = k0 + b_kofs(v, h);
    f[2 * v]     = tile[n * ldt + k];
    f[2 * v + 1] = tile[n * ldt + k + 1];
  }
  return f;
}

// ---------------------------------------------------------------------------
// One-time precision/layout conversion kernels
// ---------------------------------------------------------------------------
__global__ __launch_bounds__(256) void convert_bf16_kernel(
    const float* __restrict__ src, __bf16* __restrict__ dst, long n) {
  for (long i = (long)blockIdx.x * blockDim.x + threadIdx.x; i < n;
       i += (long)gridDim.x * blockDim.x)
    dst[i] = to_bf16(src[i]);
}

__global__ __launch_bounds__(256) void add_convert_bf16_kernel(
    const float* __restrict__ a, const float* __restrict__ b,
    __bf16* __restrict__ dst, long n) {
  for (long i = (long)blockIdx.x * blockDim.x + threadIdx.x; i < n;
       i += (long)gridDim.x * blockDim.x)
    dst[i] = to_bf16(a[i] + b[i]);
}

// Wt[n][k] = bf16(W[k][n]) via LDS tile (32x32), grid (K/32, N/32)
__global__ __launch_bounds__(256) void transpose_bf16_kernel(
    const float* __restrict__ W, __bf16* __restrict__ Wt, int K, int N) {
  __shared__ float t[32][33];
  const int kb = blockIdx.x * 32, nb = blockIdx.y * 32;
  const int tx = threadIdx.x & 31, ty = threadIdx.x >> 5;  // 32 x 8
#pragma unroll
  for (int i = ty; i < 32; i += 8) t[i][tx] = W[(long)(kb + i) * N + nb + tx];
  __syncthreads();
#pragma unroll
  for (int i = ty; i < 32; i += 8)
    Wt[(long)(nb + i) * K + kb + tx] = to_bf16(t[tx][i]);
}

// ---------------------------------------------------------------------------
// GEMM:  Y[M,N] = act( X[M,K](bf16) * Wt[N,K](bf16)^T + bias[N] )
// Block tile 128x64, BK=32, 8 wave32 waves (4x2), each wave 32x32 (4 WMMA acc).
// Double-buffered LDS tiles fed by the Tensor Data Mover (one tensor_load_to_lds
// per tile, issued by wave 0; TDM hardware padding gives 80 B LDS row stride).
// ---------------------------------------------------------------------------
template <typename OutT, bool RELU>
__global__ __launch_bounds__(256) void gemm_kernel(
    const __bf16* __restrict__ X, const __bf16* __restrict__ Wt,
    const float* __restrict__ bias, OutT* __restrict__ Y,
    int M, int K, int N) {
  constexpr int BM = 128, BN = 64, BK = 32;
  constexpr int LDA = BK + 8;  // 40 elems = 80 B = TDM 16-DW interval + 4-DW pad
  constexpr int LDB = BK + 8;
  __shared__ __bf16 As[2][BM * LDA];
  __shared__ __bf16 Bs[2][BN * LDB];

  const int tid  = threadIdx.x;
  const int lane = tid & 31;
  const int wave = tid >> 5;
  const int wm   = wave >> 1;  // 0..3
  const int wn   = wave & 1;   // 0..1
  const int m0g  = blockIdx.x * BM;
  const int n0g  = blockIdx.y * BN;

  v8f vzero = {};
  v8f acc[2][2];
#pragma unroll
  for (int i = 0; i < 2; ++i)
#pragma unroll
    for (int j = 0; j < 2; ++j) acc[i][j] = vzero;

  auto stage = [&](int k0, int buf) {  // TDM: one DMA per tile, wave 0 only
    if (wave == 0) {
      tdm_load_2d(&As[buf][0], &X[(long)m0g * K + k0], BK, BM, K, 1u, 3u, 3u);
      tdm_load_2d(&Bs[buf][0], &Wt[(long)n0g * K + k0], BK, BN, K, 1u, 3u, 3u);
    }
  };

  stage(0, 0);
  for (int k0 = 0; k0 < K; k0 += BK) {
    const int buf = (k0 / BK) & 1;
    if (wave == 0) wait_tensor0();
    __syncthreads();                          // publish TDM-written tiles
    if (k0 + BK < K) stage(k0 + BK, buf ^ 1); // overlap next DMA with math

    const v16bf a0 = load_a_frag(&As[buf][0], LDA, wm * 32, 0, lane);
    const v16bf a1 = load_a_frag(&As[buf][0], LDA, wm * 32 + 16, 0, lane);
    const v16bf b0 = load_b_frag_nk(&Bs[buf][0], LDB, 0, wn * 32, lane);
    const v16bf b1 = load_b_frag_nk(&Bs[buf][0], LDB, 0, wn * 32 + 16, lane);
    acc[0][0] = wmma_bf16(a0, b0, acc[0][0]);
    acc[0][1] = wmma_bf16(a0, b1, acc[0][1]);
    acc[1][0] = wmma_bf16(a1, b0, acc[1][0]);
    acc[1][1] = wmma_bf16(a1, b1, acc[1][1]);
    __syncthreads();                          // tiles consumed before re-fill
  }

  // epilogue: bias (+relu); C/D layout: lane -> N=lane%16, VGPR v -> M=v+8*half
  const int nlane = lane & 15;
  const int half  = lane >> 4;
#pragma unroll
  for (int mt = 0; mt < 2; ++mt)
#pragma unroll
    for (int nt = 0; nt < 2; ++nt) {
      const int gn = n0g + wn * 32 + nt * 16 + nlane;
      const float bn = bias[gn];
#pragma unroll
      for (int v = 0; v < 8; ++v) {
        const int gm = m0g + wm * 32 + mt * 16 + v + half * 8;
        float val = acc[mt][nt][v] + bn;
        if (RELU) val = fmaxf(val, 0.f);
        store_val(&Y[(long)gm * N + gn], val);
      }
    }
}

// ---------------------------------------------------------------------------
// Flash attention (single head, d=512): 32 queries/WG, 32-key blocks, online
// softmax, WMMA QK^T and PV, async global->LDS K/Q staging + transposed V
// staging, double-buffered K/V.  Grid: (Nq/32, B).
// ---------------------------------------------------------------------------
__global__ __launch_bounds__(256) void attention_kernel(
    const __bf16* __restrict__ Qg, const __bf16* __restrict__ Kg,
    const __bf16* __restrict__ Vg, float* __restrict__ Og,
    int Nk, float inv_scale) {
  constexpr int QT = 32, KBK = 32, D = 512;
  constexpr int LDV = KBK + 8;          // 40 elems = 80 B rows
  __shared__ __bf16 Qs[QT][D];          // 32 KB
  __shared__ __bf16 Ks[2][KBK][D];      // 64 KB
  __shared__ __bf16 Vs[2][D][LDV];      // 80 KB, transposed: Vs[d][key]
  __shared__ float  Ss[QT][KBK + 4];
  __shared__ __bf16 Ps[QT][KBK];
  __shared__ float  rmax[QT], rsum[QT], rscale[QT], rinv[QT];

  const int tid   = threadIdx.x;
  const int lane  = tid & 31;
  const int wave  = tid >> 5;
  const int nlane = lane & 15;
  const int half  = lane >> 4;
  const long boff = (long)blockIdx.y * Nk * D;
  const int  q0   = blockIdx.x * QT;

  auto stageK = [&](int kb, int buf) {  // async: 32 rows x 64 chunks of 16 B
    for (int i = tid; i < KBK * (D / 8); i += 256) {
      const int r = i >> 6, c = i & 63;
      async_copy16(&Ks[buf][r][c * 8], &Kg[boff + (long)(kb + r) * D + c * 8]);
    }
  };
  auto stageV = [&](int kb, int buf) {  // manual transpose into Vs[d][key]
    for (int i = tid; i < KBK * D; i += 256) {
      const int d = i & (D - 1), key = i >> 9;  // consecutive tid -> consecutive d
      Vs[buf][d][key] = Vg[boff + (long)(kb + key) * D + d];
    }
  };

  // Q tile (async), first K block (async), first V block (manual)
  for (int i = tid; i < QT * (D / 8); i += 256) {
    const int r = i >> 6, c = i & 63;
    async_copy16(&Qs[r][c * 8], &Qg[boff + (long)(q0 + r) * D + c * 8]);
  }
  stageK(0, 0);
  stageV(0, 0);
  if (tid < QT) { rmax[tid] = -1e30f; rsum[tid] = 0.f; }

  // O accumulators: wave owns M-tile (wave&1), 8 N-tiles from (wave>>1)*8
  const int omt  = wave & 1;
  const int ont0 = (wave >> 1) * 8;
  v8f vzero = {};
  v8f oacc[8];
#pragma unroll
  for (int j = 0; j < 8; ++j) oacc[j] = vzero;

  for (int kb = 0; kb < Nk; kb += KBK) {
    const int buf = (kb / KBK) & 1;
    wait_async0();
    __syncthreads();
    if (kb + KBK < Nk) {  // overlap next block's copies with math
      stageK(kb + KBK, buf ^ 1);
      stageV(kb + KBK, buf ^ 1);
    }

    // S = Q K^T * inv_scale : 4 tiles of 16x16 on waves 0..3
    if (wave < 4) {
      const int smt = wave & 1;
      const int snt = wave >> 1;
      v8f sa = vzero;
#pragma unroll
      for (int kc = 0; kc < D / 32; ++kc) {
        const v16bf a = load_a_frag(&Qs[0][0], D, smt * 16, kc * 32, lane);
        const v16bf b = load_b_frag_nk(&Ks[buf][0][0], D, kc * 32, snt * 16, lane);
        sa = wmma_bf16(a, b, sa);
      }
#pragma unroll
      for (int v = 0; v < 8; ++v)
        Ss[smt * 16 + v + half * 8][snt * 16 + nlane] = sa[v] * inv_scale;
    }
    __syncthreads();

    // online softmax row update (one thread per query row)
    if (tid < QT) {
      const int r = tid;
      const float mold = rmax[r];
      float mx = mold;
      for (int c = 0; c < KBK; ++c) mx = fmaxf(mx, Ss[r][c]);
      const float sc = __expf(mold - mx);
      float sum = 0.f;
      for (int c = 0; c < KBK; ++c) {
        const float pv = __expf(Ss[r][c] - mx);
        Ps[r][c] = to_bf16(pv);
        sum += pv;
      }
      rsum[r]   = rsum[r] * sc + sum;
      rmax[r]   = mx;
      rscale[r] = sc;
    }
    __syncthreads();

    // rescale accumulators, then O += P * V
#pragma unroll
    for (int v = 0; v < 8; ++v) {
      const float f = rscale[omt * 16 + v + half * 8];
#pragma unroll
      for (int j = 0; j < 8; ++j) oacc[j][v] *= f;
    }
    {
      const v16bf a = load_a_frag(&Ps[0][0], KBK, omt * 16, 0, lane);
#pragma unroll
      for (int j = 0; j < 8; ++j) {
        // B[k=key][n=d] with Vs[d][key]: K contiguous -> ds_load_b128
        const v16bf b = load_b_frag_nk(&Vs[buf][0][0], LDV, 0, (ont0 + j) * 16, lane);
        oacc[j] = wmma_bf16(a, b, oacc[j]);
      }
    }
    __syncthreads();
  }

  if (tid < QT) rinv[tid] = 1.0f / rsum[tid];
  __syncthreads();

#pragma unroll
  for (int j = 0; j < 8; ++j) {
    const int gn = (ont0 + j) * 16 + nlane;
#pragma unroll
    for (int v = 0; v < 8; ++v) {
      const int gm = q0 + omt * 16 + v + half * 8;
      Og[boff + (long)gm * D + gn] = oacc[j][v] * rinv[omt * 16 + v + half * 8];
    }
  }
}

// ---------------------------------------------------------------------------
// out[row] = LN(attn[row] + res[row]) * gamma + beta      (F = 512, 256 thr)
// ---------------------------------------------------------------------------
__global__ __launch_bounds__(256) void residual_ln_kernel(
    const float* __restrict__ attn, const float* __restrict__ res,
    const float* __restrict__ gamma, const float* __restrict__ beta,
    float* __restrict__ out, int F) {
  const int row = blockIdx.x;
  const int tid = threadIdx.x;
  __shared__ float red[16];

  const long base = (long)row * F;
  const float v0 = attn[base + tid] + res[base + tid];
  const float v1 = attn[base + tid + 256] + res[base + tid + 256];
  float s = v0 + v1;
  float q = v0 * v0 + v1 * v1;
#pragma unroll
  for (int m = 16; m; m >>= 1) {
    s += __shfl_xor(s, m, 32);
    q += __shfl_xor(q, m, 32);
  }
  const int wave = tid >> 5, lane = tid & 31;
  if (lane == 0) { red[wave] = s; red[wave + 8] = q; }
  __syncthreads();
  if (tid == 0) {
    float ts = 0.f, tq = 0.f;
    for (int i = 0; i < 8; ++i) { ts += red[i]; tq += red[i + 8]; }
    red[0] = ts;
    red[8] = tq;
  }
  __syncthreads();
  const float mu  = red[0] / (float)F;
  const float var = red[8] / (float)F - mu * mu;
  const float w   = rsqrtf(var + 1e-3f);
  out[base + tid]       = (v0 - mu) * w * gamma[tid] + beta[tid];
  out[base + tid + 256] = (v1 - mu) * w * gamma[tid + 256] + beta[tid + 256];
}

// ---------------------------------------------------------------------------
// Host-side launcher
// ---------------------------------------------------------------------------
extern "C" void kernel_launch(void* const* d_in, const int* in_sizes, int n_in,
                              void* d_out, int out_size, void* d_ws, size_t ws_size,
                              hipStream_t stream) {
  const float* p     = (const float*)d_in[0];
  const float* s     = (const float*)d_in[1];
  const float* sa_wq = (const float*)d_in[2];
  const float* sa_bq = (const float*)d_in[3];
  const float* sa_wk = (const float*)d_in[4];
  const float* sa_bk = (const float*)d_in[5];
  const float* sa_wv = (const float*)d_in[6];
  const float* sa_bv = (const float*)d_in[7];
  const float* ca_wq = (const float*)d_in[8];
  const float* ca_bq = (const float*)d_in[9];
  const float* ca_wk = (const float*)d_in[10];
  const float* ca_bk = (const float*)d_in[11];
  const float* ca_wv = (const float*)d_in[12];
  const float* ca_bv = (const float*)d_in[13];
  const float* ln1_g = (const float*)d_in[14];
  const float* ln1_b = (const float*)d_in[15];
  const float* ln2_g = (const float*)d_in[16];
  const float* ln2_b = (const float*)d_in[17];
  const float* ff_w1 = (const float*)d_in[18];
  const float* ff_b1 = (const float*)d_in[19];
  const float* ff_w2 = (const float*)d_in[20];
  const float* ff_b2 = (const float*)d_in[21];

  const int B = 4, N = 2048, F = 512;
  const long tok = (long)B * N;   // 8192 rows
  const long mat = tok * F;       // 4,194,304 elements
  const long wel = (long)F * F;   // 262,144 elements per weight

  // outputs: (ff_out, sa_out, ca_out) concatenated
  float* ff_out = (float*)d_out;
  float* sa_out = ff_out + mat;
  float* ca_out = ff_out + 2 * mat;

  // workspace layout (all bf16 unless noted)
  char* ws = (char*)d_ws;
  __bf16* p_bf = (__bf16*)ws;
  __bf16* s_bf = p_bf + mat;
  __bf16* wt[8];                         // transposed bf16 weights [N][K]
  for (int i = 0; i < 8; ++i) wt[i] = s_bf + mat + i * wel;
  __bf16* saQ = wt[7] + wel;
  __bf16* saK = saQ + mat;
  __bf16* saV = saK + mat;
  __bf16* caQ = saV + mat;
  __bf16* caK = caQ + mat;
  __bf16* caV = caK + mat;
  float* sa_attn = (float*)(caV + mat);
  float* ca_attn = sa_attn + mat;
  __bf16* ff_in = (__bf16*)(ca_attn + mat);
  __bf16* h     = ff_in + mat;

  const dim3 blk(256);
  const dim3 cg(4096);                     // grid-stride converts
  const dim3 tg(F / 32, F / 32);           // 16 x 16 transpose tiles
  const dim3 gg((unsigned)(tok / 128), (unsigned)(F / 64));
  const dim3 ag(N / 32, B);

  // one-time conversions: activations to bf16, weights to transposed bf16
  convert_bf16_kernel<<<cg, blk, 0, stream>>>(p, p_bf, mat);
  convert_bf16_kernel<<<cg, blk, 0, stream>>>(s, s_bf, mat);
  const float* wsrc[8] = {sa_wq, sa_wk, sa_wv, ca_wq, ca_wk, ca_wv, ff_w1, ff_w2};
  for (int i = 0; i < 8; ++i)
    transpose_bf16_kernel<<<tg, blk, 0, stream>>>(wsrc[i], wt[i], F, F);

  // QKV projections (bf16 outputs feed WMMA attention)
  gemm_kernel<__bf16, false><<<gg, blk, 0, stream>>>(p_bf, wt[0], sa_bq, saQ, (int)tok, F, F);
  gemm_kernel<__bf16, false><<<gg, blk, 0, stream>>>(p_bf, wt[1], sa_bk, saK, (int)tok, F, F);
  gemm_kernel<__bf16, false><<<gg, blk, 0, stream>>>(p_bf, wt[2], sa_bv, saV, (int)tok, F, F);
  gemm_kernel<__bf16, false><<<gg, blk, 0, stream>>>(p_bf, wt[3], ca_bq, caQ, (int)tok, F, F);
  gemm_kernel<__bf16, false><<<gg, blk, 0, stream>>>(s_bf, wt[4], ca_bk, caK, (int)tok, F, F);
  gemm_kernel<__bf16, false><<<gg, blk, 0, stream>>>(s_bf, wt[5], ca_bv, caV, (int)tok, F, F);

  // flash attention (scale = sqrt(F))
  const float inv_scale = 0.044194173824159216f;  // 1/sqrt(512)
  attention_kernel<<<ag, blk, 0, stream>>>(saQ, saK, saV, sa_attn, N, inv_scale);
  attention_kernel<<<ag, blk, 0, stream>>>(caQ, caK, caV, ca_attn, N, inv_scale);

  // residual + layernorm
  residual_ln_kernel<<<dim3((unsigned)tok), blk, 0, stream>>>(sa_attn, p, ln1_g, ln1_b, sa_out, F);
  residual_ln_kernel<<<dim3((unsigned)tok), blk, 0, stream>>>(ca_attn, s, ln2_g, ln2_b, ca_out, F);

  // feed-forward: h = relu((sa_out + ca_out) @ w1 + b1);  ff_out = h @ w2 + b2
  add_convert_bf16_kernel<<<cg, blk, 0, stream>>>(sa_out, ca_out, ff_in, mat);
  gemm_kernel<__bf16, true><<<gg, blk, 0, stream>>>(ff_in, wt[6], ff_b1, h, (int)tok, F, F);
  gemm_kernel<float, false><<<gg, blk, 0, stream>>>(h, wt[7], ff_b2, ff_out, (int)tok, F, F);
}
